// AttentiveLinear_10514079941485
// MI455X (gfx1250) — matrix operand order repaired and checked by reading, not compiled, not run on hardware
//
#include <hip/hip_runtime.h>

#define IN_F   128
#define OUT_F  128
#define NROWS  8192                      // 4 * 2048
// Symmetry-folded K-steps: sum_k (4 - k/32) = 320 triangular steps + 4 bias steps
#define NSTEPS 324
#define W2_ELEMS ((size_t)NSTEPS * 8 * 512)   // 1,327,104 bf16 (~2.53 MiB)

typedef __attribute__((ext_vector_type(16))) __bf16 v16bf;
typedef __attribute__((ext_vector_type(8)))  float  v8f;

// f32 -> bf16 round-to-nearest-even (finite inputs)
static __device__ __forceinline__ unsigned short f2bf_rne(float f) {
    unsigned u = __builtin_bit_cast(unsigned, f);
    return (unsigned short)((u + 0x7FFFu + ((u >> 16) & 1u)) >> 16);
}

// packed bf16 multiply (VOP3P op 42); non-volatile so the scheduler can interleave
static __device__ __forceinline__ unsigned pk_mul_bf16(unsigned a, unsigned b) {
    unsigned d;
    asm("v_pk_mul_bf16 %0, %1, %2" : "=v"(d) : "v"(a), "v"(b));
    return d;
}

static __device__ __forceinline__ unsigned splat_bf16(unsigned short v) {
    unsigned u = v;
    return u | (u << 16);
}

// ---------------- Prep 1: x (f32) -> xb (bf16), row-major [8192][128] ----------------
__global__ __launch_bounds__(256) void cvt_x_kernel(const float* __restrict__ x,
                                                    unsigned short* __restrict__ xb) {
    int i = blockIdx.x * 256 + threadIdx.x;
    xb[i] = f2bf_rne(x[i]);
}

// ---------------- Prep 2: symmetry-folded W2, pre-swizzled into B-fragment order --------
// Step order (must match main kernel): k=0..31: ib=0..3 | k=32..63: ib=1..3 |
// k=64..95: ib=2..3 | k=96..127: ib=3 | bias: ib=0..3.
// Tile (t, c): w2s[((t*8 + c)*32 + lane)*16 + e]; lane={half=(koff>>3)&1, r=o%16},
// koff = (e<8 ? half*8 + e : 16 + half*8 + e-8) in [0,32), i = ib*32 + koff, o = c*16 + r.
// Rows: i<k -> 0 (masks the partial block), i==k -> Ww[k,k,o] (diagonal, x_k^2 term),
//       i>k -> Ww[k,i,o] + Ww[i,k,o] (folded), bias -> Wb + bw.reshape(128,128).
__global__ __launch_bounds__(256) void build_w2_kernel(const float* __restrict__ Ww,
                                                       const float* __restrict__ Wb,
                                                       const float* __restrict__ bw,
                                                       unsigned short* __restrict__ w2s) {
    int idx  = blockIdx.x * 256 + threadIdx.x;    // < NSTEPS*8*512
    int e    = idx & 15;
    int lane = (idx >> 4) & 31;
    int tile = idx >> 9;
    int c    = tile & 7;
    int t    = tile >> 3;
    int o    = c * 16 + (lane & 15);
    int koff = ((e >> 3) << 4) + ((lane >> 4) << 3) + (e & 7); // 0..31 within the K-step
    float v;
    if (t >= 320) {                               // folded bias GEMM rows
        int i2 = (t - 320) * 32 + koff;
        v = Wb[i2 * OUT_F + o] + bw[i2 * OUT_F + o];
    } else {
        int k, ib;
        if (t < 128)      { k = t >> 2;                 ib = t & 3; }
        else if (t < 224) { int u = t - 128; k = 32 + u / 3;    ib = 1 + u % 3; }
        else if (t < 288) { int u = t - 224; k = 64 + (u >> 1); ib = 2 + (u & 1); }
        else              { k = 96 + (t - 288);         ib = 3; }
        int i = ib * 32 + koff;
        if (i < k) {
            v = 0.f;                                                    // mask lower triangle
        } else if (i == k) {
            v = Ww[(size_t)k * 16384 + (size_t)i * 128 + o];            // diagonal
        } else {
            v = Ww[(size_t)k * 16384 + (size_t)i * 128 + o]
              + Ww[(size_t)i * 16384 + (size_t)k * 128 + o];            // symmetric fold
        }
    }
    w2s[idx] = f2bf_rne(v);
}

// ---------------- Main: y = P_sym @ W2fold + bb, P generated on the fly ----------------
// Block: 64 rows x 128 cols, 8 waves. Wave tile: 32x32 (2 M-tiles x 2 N-tiles).
__global__ __launch_bounds__(256) void attn_gemm_kernel(const unsigned short* __restrict__ xb,
                                                        const unsigned short* __restrict__ w2s,
                                                        const float* __restrict__ bb,
                                                        float* __restrict__ out) {
    // 64 rows x 128 bf16, padded to 136 shorts (272 B) per row for LDS bank spread
    __shared__ unsigned short xt[64 * 136];

    const int tid  = threadIdx.x;
    const int lane = tid & 31;
    const int wave = tid >> 5;
    const int mg   = wave >> 2;      // 0..1 : row group
    const int cg   = wave & 3;       // 0..3 : column group (32 cols)
    const int n0   = blockIdx.x * 64;

    for (int ck = tid; ck < 1024; ck += 256) {
        int row = ck >> 4, c8 = ck & 15;
        uint4 v = *(const uint4*)(xb + (size_t)(n0 + row) * IN_F + c8 * 8);
        *(uint4*)(&xt[row * 136 + c8 * 8]) = v;
    }
    __syncthreads();

    const int half = lane >> 4;
    const int r16  = lane & 15;

    v8f acc00 = {0.f,0.f,0.f,0.f,0.f,0.f,0.f,0.f};
    v8f acc01 = {0.f,0.f,0.f,0.f,0.f,0.f,0.f,0.f};
    v8f acc10 = {0.f,0.f,0.f,0.f,0.f,0.f,0.f,0.f};
    v8f acc11 = {0.f,0.f,0.f,0.f,0.f,0.f,0.f,0.f};

    const unsigned short* xrow0 = &xt[(mg * 32 + r16) * 136];       // M-tile 0 row
    const unsigned short* xrow1 = xrow0 + 16 * 136;                 // M-tile 1 row
    const uint4* bptr = (const uint4*)(w2s + ((size_t)(cg * 2) * 32 + lane) * 16);

    union Frag { uint4 q[2]; v16bf v; };

    // One K-step: A = s * x[i-block], 4 WMMAs over the 2x2 wave tile.
    auto step = [&](unsigned s0, unsigned s1, int ib) {
        Frag A0, A1, B0, B1;
        B0.q[0] = bptr[0];   B0.q[1] = bptr[1];      // N-tile 0
        B1.q[0] = bptr[64];  B1.q[1] = bptr[65];     // N-tile 1 (+1 KiB)
        __builtin_prefetch(bptr + 512, 0, 1);        // next 8 KiB slab
        const int boff = ib * 64 + half * 16;
        uint4 a0 = *(const uint4*)((const char*)xrow0 + boff);
        uint4 a1 = *(const uint4*)((const char*)xrow0 + boff + 32);
        uint4 c0 = *(const uint4*)((const char*)xrow1 + boff);
        uint4 c1 = *(const uint4*)((const char*)xrow1 + boff + 32);
        A0.q[0].x = pk_mul_bf16(a0.x, s0); A0.q[0].y = pk_mul_bf16(a0.y, s0);
        A0.q[0].z = pk_mul_bf16(a0.z, s0); A0.q[0].w = pk_mul_bf16(a0.w, s0);
        A0.q[1].x = pk_mul_bf16(a1.x, s0); A0.q[1].y = pk_mul_bf16(a1.y, s0);
        A0.q[1].z = pk_mul_bf16(a1.z, s0); A0.q[1].w = pk_mul_bf16(a1.w, s0);
        A1.q[0].x = pk_mul_bf16(c0.x, s1); A1.q[0].y = pk_mul_bf16(c0.y, s1);
        A1.q[0].z = pk_mul_bf16(c0.z, s1); A1.q[0].w = pk_mul_bf16(c0.w, s1);
        A1.q[1].x = pk_mul_bf16(c1.x, s1); A1.q[1].y = pk_mul_bf16(c1.y, s1);
        A1.q[1].z = pk_mul_bf16(c1.z, s1); A1.q[1].w = pk_mul_bf16(c1.w, s1);
        acc00 = __builtin_amdgcn_wmma_f32_16x16x32_bf16(false, A0.v, false, B0.v,
                                                        (short)0, acc00, false, false);
        acc01 = __builtin_amdgcn_wmma_f32_16x16x32_bf16(false, A0.v, false, B1.v,
                                                        (short)0, acc01, false, false);
        acc10 = __builtin_amdgcn_wmma_f32_16x16x32_bf16(false, A1.v, false, B0.v,
                                                        (short)0, acc10, false, false);
        acc11 = __builtin_amdgcn_wmma_f32_16x16x32_bf16(false, A1.v, false, B1.v,
                                                        (short)0, acc11, false, false);
        bptr += 512;
    };

    // Bias step: scale == 1, feed raw x fragment (no pk_mul needed).
    auto step_id = [&](int ib) {
        Frag A0, A1, B0, B1;
        B0.q[0] = bptr[0];   B0.q[1] = bptr[1];
        B1.q[0] = bptr[64];  B1.q[1] = bptr[65];
        const int boff = ib * 64 + half * 16;
        A0.q[0] = *(const uint4*)((const char*)xrow0 + boff);
        A0.q[1] = *(const uint4*)((const char*)xrow0 + boff + 32);
        A1.q[0] = *(const uint4*)((const char*)xrow1 + boff);
        A1.q[1] = *(const uint4*)((const char*)xrow1 + boff + 32);
        acc00 = __builtin_amdgcn_wmma_f32_16x16x32_bf16(false, A0.v, false, B0.v,
                                                        (short)0, acc00, false, false);
        acc01 = __builtin_amdgcn_wmma_f32_16x16x32_bf16(false, A0.v, false, B1.v,
                                                        (short)0, acc01, false, false);
        acc10 = __builtin_amdgcn_wmma_f32_16x16x32_bf16(false, A1.v, false, B0.v,
                                                        (short)0, acc10, false, false);
        acc11 = __builtin_amdgcn_wmma_f32_16x16x32_bf16(false, A1.v, false, B1.v,
                                                        (short)0, acc11, false, false);
        bptr += 512;
    };

    // Upper-triangle visitation (order matches build_w2_kernel exactly)
    for (int k = 0; k < 32; ++k) {
        unsigned s0 = splat_bf16(xrow0[k]), s1 = splat_bf16(xrow1[k]);
        step(s0, s1, 0); step(s0, s1, 1); step(s0, s1, 2); step(s0, s1, 3);
    }
    for (int k = 32; k < 64; ++k) {
        unsigned s0 = splat_bf16(xrow0[k]), s1 = splat_bf16(xrow1[k]);
        step(s0, s1, 1); step(s0, s1, 2); step(s0, s1, 3);
    }
    for (int k = 64; k < 96; ++k) {
        unsigned s0 = splat_bf16(xrow0[k]), s1 = splat_bf16(xrow1[k]);
        step(s0, s1, 2); step(s0, s1, 3);
    }
    for (int k = 96; k < 128; ++k) {
        unsigned s0 = splat_bf16(xrow0[k]), s1 = splat_bf16(xrow1[k]);
        step(s0, s1, 3);
    }
    step_id(0); step_id(1); step_id(2); step_id(3);      // folded bias GEMM

    // Epilogue: D layout lane={N=r16, half}, VGPR r -> M = r + 8*half
    const int o0 = cg * 32 + r16;
    const int o1 = o0 + 16;
    const float bb0 = bb[o0];
    const float bb1 = bb[o1];
    float* outp = out + (size_t)(n0 + mg * 32 + 8 * half) * OUT_F;
#pragma unroll
    for (int r = 0; r < 8; ++r) {
        outp[(size_t)r * OUT_F + o0]        = acc00[r] + bb0;
        outp[(size_t)r * OUT_F + o1]        = acc01[r] + bb1;
        outp[(size_t)(16 + r) * OUT_F + o0] = acc10[r] + bb0;
        outp[(size_t)(16 + r) * OUT_F + o1] = acc11[r] + bb1;
    }
}

extern "C" void kernel_launch(void* const* d_in, const int* in_sizes, int n_in,
                              void* d_out, int out_size, void* d_ws, size_t ws_size,
                              hipStream_t stream) {
    const float* x  = (const float*)d_in[0];   // [8192,128]
    const float* Wb = (const float*)d_in[1];   // [128,128]
    const float* bb = (const float*)d_in[2];   // [128]
    const float* Ww = (const float*)d_in[3];   // [128,16384]
    const float* bw = (const float*)d_in[4];   // [16384]

    unsigned short* xb  = (unsigned short*)d_ws;                                     // 2 MiB
    unsigned short* w2s = (unsigned short*)((char*)d_ws + (size_t)NROWS * IN_F * 2); // ~2.53 MiB

    cvt_x_kernel<<<(NROWS * IN_F) / 256, 256, 0, stream>>>(x, xb);
    build_w2_kernel<<<(int)(W2_ELEMS / 256), 256, 0, stream>>>(Ww, Wb, bw, w2s);
    attn_gemm_kernel<<<NROWS / 64, 256, 0, stream>>>(xb, w2s, bb, (float*)d_out);
}
